// PolymerGNN_SchNet_IV_87471303950764
// MI455X (gfx1250) — matrix operand order repaired and checked by reading, not compiled, not run on hardware
//
#include <hip/hip_runtime.h>
#include <hip/hip_bf16.h>

#define N_ATOMS 16384
#define N_EDGES 524288
#define NGRAPHS 256
#define NGAUSS  50
#define CUTOFF  10.0f
#define LDW     72           // LDS row pitch in halfs (64 + 8 pad, keeps 16B row align: 144B)

typedef __attribute__((ext_vector_type(16))) _Float16 v16h;
typedef __attribute__((ext_vector_type(8)))  _Float16 v8h;   // 16B, natural align 16
typedef __attribute__((ext_vector_type(8)))  float    v8f;

// ---------------------------------------------------------------------------
// Permuted A layout: within each 32-wide K block, columns stored in order
// [0-7, 16-23, 8-15, 24-31] so each lane's WMMA A-fragment (two 8-half runs)
// becomes 16 CONTIGUOUS halfs -> 2x ds_load_b128.
//   position(k' in 0..31) = 16*((k'>>3)&1) + (k'&7) + ((k'&16) ? 8 : 0)
//   inverse(pos p)        : g2=p>>4, w=p&15 ; k' = (w<8) ? 8*g2+w : 16+8*g2+(w-8)
// ---------------------------------------------------------------------------
__device__ __forceinline__ int aperm(int c) {        // full column 0..63 -> position
  const int kk = c & 32, k2 = c & 31;
  const int g = (k2 >> 3) & 1;
  const int w = (k2 & 7) + ((k2 & 16) ? 8 : 0);
  return kk + 16 * g + w;
}

__device__ __forceinline__ v16h ld_frag32(const _Float16* p) { // 16 contiguous halfs
  v8h a = *(const v8h*)p;
  v8h b = *(const v8h*)(p + 8);
  return __builtin_shufflevector(a, b, 0, 1, 2, 3, 4, 5, 6, 7,
                                 8, 9, 10, 11, 12, 13, 14, 15);
}

// A fragment from permuted-layout LDS tile (16 rows x 64 pos, pitch lda)
__device__ __forceinline__ v16h frag_a(const _Float16* A, int lda, int koff) {
  const int lane = threadIdx.x & 31;
  const int m = lane & 15, g = lane >> 4;
  return ld_frag32(A + m * lda + koff + 16 * g);
}

// B fragment from TRANSPOSED weights in LDS: Bt[col*ldb + k]
__device__ __forceinline__ v16h frag_b(const _Float16* Bt, int ldb, int koff, int noff) {
  const int lane = threadIdx.x & 31;
  const int n = lane & 15, g = lane >> 4;
  const _Float16* p = Bt + (noff + n) * ldb + koff + 8 * g;
  v8h lo = *(const v8h*)p;
  v8h hi = *(const v8h*)(p + 16);
  return __builtin_shufflevector(lo, hi, 0, 1, 2, 3, 4, 5, 6, 7,
                                 8, 9, 10, 11, 12, 13, 14, 15);
}

// D[16,64] += A[16,64] * B[64,64]   (two K=32 steps, four 16-col tiles)
__device__ __forceinline__ void gemm_16x64x64(const _Float16* A, const _Float16* Bt,
                                              v8f acc[4]) {
#pragma unroll
  for (int kk = 0; kk < 64; kk += 32) {
    v16h a = frag_a(A, LDW, kk);
#pragma unroll
    for (int nt = 0; nt < 4; ++nt) {
      v16h b = frag_b(Bt, LDW, kk, nt * 16);
      acc[nt] = __builtin_amdgcn_wmma_f32_16x16x32_f16(
          false, a, false, b, (short)0, acc[nt], false, false);
    }
  }
}

__device__ __forceinline__ float ssp(float x) {  // shifted softplus
  float sp = (x > 20.f) ? x : __logf(1.f + __expf(x));
  return sp - 0.69314718f;
}

// Stage one 16x64 f32 row-major source into permuted f16 LDS tile.
// Lane (r = lane&15) handles K-block (kk = 32*(lane>>4)): 32 halfs contiguous.
__device__ __forceinline__ void stage_a_f32(const float* __restrict__ src,
                                            _Float16* __restrict__ A) {
  const int lane = threadIdx.x & 31;
  const int r = lane & 15, kk = (lane >> 4) * 32;
  v8h buf[4];
#pragma unroll
  for (int q = 0; q < 4; ++q) {
#pragma unroll
    for (int w8 = 0; w8 < 8; ++w8) {
      const int p = q * 8 + w8;
      const int g2 = p >> 4, w = p & 15;
      const int k2 = (w < 8) ? (8 * g2 + w) : (16 + 8 * g2 + (w - 8));
      buf[q][w8] = (_Float16)src[r * 64 + kk + k2];
    }
    *(v8h*)(A + r * LDW + kk + q * 8) = buf[q];
  }
}

// ---------------------------------------------------------------------------
// h = emb[z]
// ---------------------------------------------------------------------------
__global__ void embed_kernel(const int* __restrict__ z, const float* __restrict__ emb,
                             float* __restrict__ h) {
  int i = blockIdx.x * blockDim.x + threadIdx.x;   // over N*64
  int n = i >> 6, c = i & 63;
  h[i] = emb[z[n] * 64 + c];
}

// ---------------------------------------------------------------------------
// per-edge distance + cosine cutoff
// ---------------------------------------------------------------------------
__global__ void edge_geom_kernel(const float* __restrict__ pos, const int* __restrict__ edge,
                                 float* __restrict__ d, float* __restrict__ C) {
  int e = blockIdx.x * blockDim.x + threadIdx.x;
  if (e >= N_EDGES) return;
  int s = edge[e], t = edge[N_EDGES + e];
  float dx = pos[s * 3 + 0] - pos[t * 3 + 0];
  float dy = pos[s * 3 + 1] - pos[t * 3 + 1];
  float dz = pos[s * 3 + 2] - pos[t * 3 + 2];
  float dd = sqrtf(dx * dx + dy * dy + dz * dz + 1e-12f);
  d[e] = dd;
  C[e] = 0.5f * (cosf(dd * 0.31415926535f) + 1.f);
}

// ---------------------------------------------------------------------------
// x = h @ w   (no bias), WMMA over 16-row tiles
// ---------------------------------------------------------------------------
__global__ void lin_fwd_kernel(const float* __restrict__ hin, const float* __restrict__ w,
                               float* __restrict__ out, int tilesPerWave) {
  __shared__ _Float16 sWt[64 * LDW];          // transposed: [col][k]
  __shared__ _Float16 sA[8][16 * LDW];
  const int tid = threadIdx.x;
  for (int p = tid; p < 4096; p += blockDim.x) {
    int k = p >> 6, c = p & 63;
    sWt[c * LDW + k] = (_Float16)w[p];
  }
  __syncthreads();

  const int wave = tid >> 5, lane = tid & 31, g = lane >> 4;
  _Float16* A = sA[wave];
  const int waveId = blockIdx.x * 8 + wave;

  for (int t = 0; t < tilesPerWave; ++t) {
    int tile = waveId * tilesPerWave + t;
    int row0 = tile * 16;
    if (row0 >= N_ATOMS) break;
    stage_a_f32(hin + (size_t)row0 * 64, A);
    __builtin_amdgcn_wave_barrier();
    v8f acc[4] = {};
    gemm_16x64x64(A, sWt, acc);
#pragma unroll
    for (int nt = 0; nt < 4; ++nt) {
      int col = nt * 16 + (lane & 15);
#pragma unroll
      for (int r = 0; r < 8; ++r)
        out[(row0 + r + 8 * g) * 64 + col] = acc[nt][r];
    }
    __builtin_amdgcn_wave_barrier();
  }
}

// ---------------------------------------------------------------------------
// Fused edge kernel: RBF -> ssp(ea@w1+b1)@w2+b2 -> *C -> gather x[src] ->
// atomic scatter into agg[dst].  One 16-edge tile per wave iteration.
// ---------------------------------------------------------------------------
__global__ void edge_filter_agg_kernel(const float* __restrict__ d, const float* __restrict__ Cc,
                                       const int* __restrict__ edge,
                                       const float* __restrict__ x, float* __restrict__ agg,
                                       const float* __restrict__ w1, const float* __restrict__ b1,
                                       const float* __restrict__ w2, const float* __restrict__ b2,
                                       int tilesPerWave) {
  __shared__ _Float16 sW1t[64 * LDW];         // transposed [col][k], rows >= NGAUSS zero
  __shared__ _Float16 sW2t[64 * LDW];
  __shared__ float sB1[64], sB2[64];
  __shared__ _Float16 sA[8][16 * LDW];
  __shared__ _Float16 sT[8][16 * LDW];
  __shared__ float sC[8][16];
  __shared__ int sSrc[8][16], sDst[8][16];

  const int tid = threadIdx.x;
  for (int p = tid; p < 4096; p += blockDim.x) {
    int k = p >> 6, c = p & 63;
    sW1t[c * LDW + k] = (_Float16)((k < NGAUSS) ? w1[k * 64 + c] : 0.f);
    sW2t[c * LDW + k] = (_Float16)w2[p];
  }
  if (tid < 64) { sB1[tid] = b1[tid]; sB2[tid] = b2[tid]; }
  __syncthreads();

  const int wave = tid >> 5, lane = tid & 31, g = lane >> 4;
  _Float16* A = sA[wave];
  _Float16* T = sT[wave];
  const int waveId = blockIdx.x * 8 + wave;

  const float delta = CUTOFF / (NGAUSS - 1);
  const float coeff = -0.5f / (delta * delta);

  for (int t = 0; t < tilesPerWave; ++t) {
    int tile = waveId * tilesPerWave + t;
    int eb = tile * 16;
    if (eb >= N_EDGES) break;

    if (lane < 16) {
      int e = eb + lane;
      sSrc[wave][lane] = edge[e];
      sDst[wave][lane] = edge[N_EDGES + e];
      sC[wave][lane] = Cc[e];
    }
    // Gaussian RBF tile: lane (r, K-block kk) emits 32 permuted halfs, 4x b128 stores
    {
      const int r = lane & 15, kk = (lane >> 4) * 32;
      const float de = d[eb + r];
      v8h buf[4];
#pragma unroll
      for (int q = 0; q < 4; ++q) {
#pragma unroll
        for (int w8 = 0; w8 < 8; ++w8) {
          const int p = q * 8 + w8;
          const int g2 = p >> 4, w = p & 15;
          const int k2 = (w < 8) ? (8 * g2 + w) : (16 + 8 * g2 + (w - 8));
          const int c = kk + k2;
          float v = 0.f;
          if (c < NGAUSS) {
            float u = de - (float)c * delta;
            v = __expf(coeff * u * u);
          }
          buf[q][w8] = (_Float16)v;
        }
        *(v8h*)(A + r * LDW + kk + q * 8) = buf[q];
      }
    }
    __builtin_amdgcn_wave_barrier();

    v8f acc[4] = {};
    gemm_16x64x64(A, sW1t, acc);            // ea @ w1

#pragma unroll
    for (int nt = 0; nt < 4; ++nt) {
      int col = nt * 16 + (lane & 15);
      float bb = sB1[col];
      int pc = aperm(col);
#pragma unroll
      for (int r = 0; r < 8; ++r) {
        int row = r + 8 * g;
        T[row * LDW + pc] = (_Float16)ssp(acc[nt][r] + bb);
        acc[nt][r] = 0.f;
      }
    }
    __builtin_amdgcn_wave_barrier();

    gemm_16x64x64(T, sW2t, acc);            // ssp(.) @ w2

#pragma unroll
    for (int nt = 0; nt < 4; ++nt) {
      int col = nt * 16 + (lane & 15);
      float bb = sB2[col];
#pragma unroll
      for (int r = 0; r < 8; ++r) {
        int row = r + 8 * g;
        float wv = (acc[nt][r] + bb) * sC[wave][row];
        int s = sSrc[wave][row];
        int dn = sDst[wave][row];
        float xv = x[s * 64 + col];
        unsafeAtomicAdd(&agg[dn * 64 + col], xv * wv);
      }
    }
    __builtin_amdgcn_wave_barrier();
  }
}

// ---------------------------------------------------------------------------
// h += ssp(agg @ w2 + b2) @ w3 + b3
// ---------------------------------------------------------------------------
__global__ void update_h_kernel(float* __restrict__ h, const float* __restrict__ agg,
                                const float* __restrict__ w2, const float* __restrict__ b2,
                                const float* __restrict__ w3, const float* __restrict__ b3,
                                int tilesPerWave) {
  __shared__ _Float16 sW2t[64 * LDW], sW3t[64 * LDW];   // transposed [col][k]
  __shared__ float sB2[64], sB3[64];
  __shared__ _Float16 sA[8][16 * LDW], sT[8][16 * LDW];
  const int tid = threadIdx.x;
  for (int p = tid; p < 4096; p += blockDim.x) {
    int k = p >> 6, c = p & 63;
    sW2t[c * LDW + k] = (_Float16)w2[p];
    sW3t[c * LDW + k] = (_Float16)w3[p];
  }
  if (tid < 64) { sB2[tid] = b2[tid]; sB3[tid] = b3[tid]; }
  __syncthreads();

  const int wave = tid >> 5, lane = tid & 31, g = lane >> 4;
  _Float16* A = sA[wave];
  _Float16* T = sT[wave];
  const int waveId = blockIdx.x * 8 + wave;

  for (int t = 0; t < tilesPerWave; ++t) {
    int tile = waveId * tilesPerWave + t;
    int row0 = tile * 16;
    if (row0 >= N_ATOMS) break;
    stage_a_f32(agg + (size_t)row0 * 64, A);
    __builtin_amdgcn_wave_barrier();

    v8f acc[4] = {};
    gemm_16x64x64(A, sW2t, acc);
#pragma unroll
    for (int nt = 0; nt < 4; ++nt) {
      int col = nt * 16 + (lane & 15);
      float bb = sB2[col];
      int pc = aperm(col);
#pragma unroll
      for (int r = 0; r < 8; ++r) {
        T[(r + 8 * g) * LDW + pc] = (_Float16)ssp(acc[nt][r] + bb);
        acc[nt][r] = 0.f;
      }
    }
    __builtin_amdgcn_wave_barrier();

    gemm_16x64x64(T, sW3t, acc);
#pragma unroll
    for (int nt = 0; nt < 4; ++nt) {
      int col = nt * 16 + (lane & 15);
      float bb = sB3[col];
#pragma unroll
      for (int r = 0; r < 8; ++r) {
        int row = row0 + r + 8 * g;
        h[row * 64 + col] += acc[nt][r] + bb;
      }
    }
    __builtin_amdgcn_wave_barrier();
  }
}

// ---------------------------------------------------------------------------
// readout: mean over graphs of segment_sum == sum(h)/NGRAPHS (every atom
// belongs to exactly one graph) -> column sums scaled by 1/256.
// ---------------------------------------------------------------------------
__global__ void pool_mean_kernel(const float* __restrict__ h, float* __restrict__ epool) {
  const int tid = threadIdx.x;
  const int c = tid & 63;
  const int rg = blockIdx.x * (blockDim.x >> 6) + (tid >> 6);
  const int stride = gridDim.x * (blockDim.x >> 6);
  float s = 0.f;
  for (int n = rg; n < N_ATOMS; n += stride) s += h[n * 64 + c];
  unsafeAtomicAdd(&epool[c], s * (1.f / NGRAPHS));
}

// ---------------------------------------------------------------------------
// head MLP: concat(eA,eG,add) @ fc1 + b -> PReLU -> @ fc2 + b -> exp
// ---------------------------------------------------------------------------
__global__ void head_kernel(const float* __restrict__ epool, const float* __restrict__ addf,
                            const float* __restrict__ fc1w, const float* __restrict__ fc1b,
                            const float* __restrict__ alpha, const float* __restrict__ fc2w,
                            const float* __restrict__ fc2b, float* __restrict__ out) {
  __shared__ float pool[130];
  __shared__ float red[64];
  const int t = threadIdx.x;   // 64 threads
  for (int i = t; i < 128; i += 64) pool[i] = epool[i];
  if (t < 2) pool[128 + t] = addf[t];
  __syncthreads();
  float acc = fc1b[t];
  for (int r = 0; r < 130; ++r) acc += pool[r] * fc1w[r * 64 + t];
  float a = alpha[0];
  acc = (acc >= 0.f) ? acc : a * acc;
  red[t] = acc * fc2w[t];
  __syncthreads();
  for (int s = 32; s > 0; s >>= 1) {
    if (t < s) red[t] += red[t + s];
    __syncthreads();
  }
  if (t == 0) out[0] = __expf(red[0] + fc2b[0]);
}

// ---------------------------------------------------------------------------
extern "C" void kernel_launch(void* const* d_in, const int* in_sizes, int n_in,
                              void* d_out, int out_size, void* d_ws, size_t ws_size,
                              hipStream_t stream) {
  const int*   zA    = (const int*)d_in[0];
  const float* posA  = (const float*)d_in[1];
  const int*   edgeA = (const int*)d_in[3];
  const int*   zG    = (const int*)d_in[4];
  const float* posG  = (const float*)d_in[5];
  const int*   edgeG = (const int*)d_in[7];
  const float* addf  = (const float*)d_in[8];
  const float* emb   = (const float*)d_in[9];
  const float* mw1   = (const float*)d_in[10];
  const float* mb1   = (const float*)d_in[11];
  const float* mw2   = (const float*)d_in[12];
  const float* mb2   = (const float*)d_in[13];
  const float* l1w   = (const float*)d_in[14];
  const float* l2w   = (const float*)d_in[15];
  const float* l2b   = (const float*)d_in[16];
  const float* l3w   = (const float*)d_in[17];
  const float* l3b   = (const float*)d_in[18];
  const float* fc1w  = (const float*)d_in[19];
  const float* fc1b  = (const float*)d_in[20];
  const float* alpha = (const float*)d_in[21];
  const float* fc2w  = (const float*)d_in[22];
  const float* fc2b  = (const float*)d_in[23];

  // workspace layout (floats)
  float* ws   = (float*)d_ws;
  const size_t NH = (size_t)N_ATOMS * 64;
  float* hA   = ws;            float* hG   = hA + NH;
  float* xA   = hG + NH;       float* xG   = xA + NH;
  float* aggA = xG + NH;       float* aggG = aggA + NH;
  float* dA   = aggG + NH;     float* CA   = dA + N_EDGES;
  float* dG   = CA + N_EDGES;  float* CG   = dG + N_EDGES;
  float* epool = CG + N_EDGES; // 128 floats: [eA(64), eG(64)]

  // --- embed + edge geometry -------------------------------------------------
  embed_kernel<<<(N_ATOMS * 64) / 256, 256, 0, stream>>>(zA, emb, hA);
  embed_kernel<<<(N_ATOMS * 64) / 256, 256, 0, stream>>>(zG, emb, hG);
  edge_geom_kernel<<<N_EDGES / 256, 256, 0, stream>>>(posA, edgeA, dA, CA);
  edge_geom_kernel<<<N_EDGES / 256, 256, 0, stream>>>(posG, edgeG, dG, CG);

  // --- 6 interaction blocks for each graph ----------------------------------
  const float* dArr[2]   = { dA, dG };
  const float* CArr[2]   = { CA, CG };
  const int*   eArr[2]   = { edgeA, edgeG };
  float*       hArr[2]   = { hA, hG };
  float*       xArr[2]   = { xA, xG };
  float*       gArr[2]   = { aggA, aggG };

  for (int i = 0; i < 6; ++i) {
    const float* w1  = mw1 + (size_t)i * NGAUSS * 64;
    const float* b1  = mb1 + (size_t)i * 64;
    const float* w2  = mw2 + (size_t)i * 64 * 64;
    const float* b2  = mb2 + (size_t)i * 64;
    const float* wl1 = l1w + (size_t)i * 64 * 64;
    const float* wl2 = l2w + (size_t)i * 64 * 64;
    const float* bl2 = l2b + (size_t)i * 64;
    const float* wl3 = l3w + (size_t)i * 64 * 64;
    const float* bl3 = l3b + (size_t)i * 64;

    for (int gidx = 0; gidx < 2; ++gidx) {
      // x = h @ lin1   (1024 row tiles; 64 blocks * 8 waves * 2 tiles)
      lin_fwd_kernel<<<64, 256, 0, stream>>>(hArr[gidx], wl1, xArr[gidx], 2);
      hipMemsetAsync(gArr[gidx], 0, NH * sizeof(float), stream);
      // fused filter + gather/scatter (32768 edge tiles; 512 blocks * 8 waves * 8)
      edge_filter_agg_kernel<<<512, 256, 0, stream>>>(
          dArr[gidx], CArr[gidx], eArr[gidx], xArr[gidx], gArr[gidx],
          w1, b1, w2, b2, 8);
      // h += mlp(agg)
      update_h_kernel<<<64, 256, 0, stream>>>(hArr[gidx], gArr[gidx],
                                              wl2, bl2, wl3, bl3, 2);
    }
  }

  // --- readout + head --------------------------------------------------------
  hipMemsetAsync(epool, 0, 128 * sizeof(float), stream);
  pool_mean_kernel<<<64, 256, 0, stream>>>(hA, epool);
  pool_mean_kernel<<<64, 256, 0, stream>>>(hG, epool + 64);
  head_kernel<<<1, 64, 0, stream>>>(epool, addf, fc1w, fc1b, alpha, fc2w, fc2b,
                                    (float*)d_out);
}
